// Block_36962488549943
// MI455X (gfx1250) — matrix operand order repaired and checked by reading, not compiled, not run on hardware
//
#include <hip/hip_runtime.h>

#define B_  32
#define T_  768
#define C_  256
#define H_  4
#define HS  64
#define F_  1024
#define BT_ (B_*T_)

typedef __attribute__((ext_vector_type(16))) __bf16   v16bf;
typedef __attribute__((ext_vector_type(8)))  float    v8f;
typedef __attribute__((ext_vector_type(4)))  unsigned u32x4;
typedef __attribute__((ext_vector_type(8)))  unsigned u32x8;

union AFrag { v16bf v; unsigned u[8]; };
union BFrag { v16bf v; u32x4 q[2]; unsigned u[8]; };

// ---- fragment loaders ------------------------------------------------------
// A (16x32 bf16, MxK): lane holds row m=lane&15; K layout used here (same as B):
//   VGPR v : K = 2v + p + 16*half (16 contiguous K per lane)
__device__ __forceinline__ AFrag load_a(const __bf16* a, int lda, int lane) {
  AFrag f;
  int m = lane & 15, half = lane >> 4;
  const unsigned* base = (const unsigned*)(a + (size_t)m * lda);
  int o = 8 * half;
#pragma unroll
  for (int vv = 0; vv < 4; ++vv) f.u[vv] = base[o + vv];
#pragma unroll
  for (int vv = 0; vv < 4; ++vv) f.u[4 + vv] = base[4 + o + vv];
  return f;
}

// B (32x16 bf16, KxN) where WMMA element (k,n) lives at bsrc[n*ldb + k]:
// lane holds col n=lane&15; K = 2v + p + 16*half  -> 16 contiguous bf16/lane.
__device__ __forceinline__ BFrag load_bT(const __bf16* bsrc, int ldb, int lane) {
  BFrag f;
  int n = lane & 15, half = lane >> 4;
  const u32x4* base = (const u32x4*)(bsrc + (size_t)n * ldb + 16 * half);
  f.q[0] = base[0];
  f.q[1] = base[1];
  return f;
}

__device__ __forceinline__ v8f wmma_bf16(const AFrag& a, const BFrag& b, v8f c) {
  return __builtin_amdgcn_wmma_f32_16x16x32_bf16(false, a.v, false, b.v,
                                                 (short)0, c, false, false);
}

// ---- Tensor Data Mover: DMA a 16-row A tile (rowBytes each) into LDS -------
// D# built per CDNA5 ISA: 2D tile, data_size = 8 bytes.
// Issued once per block (by wave 0); tracked with TENSORcnt.
__device__ __forceinline__ void tdm_load_tile16(const void* gsrc, unsigned ldsAddr,
                                                unsigned rowBytes, unsigned strideBytes) {
  unsigned long long ga = (unsigned long long)(size_t)gsrc;
  unsigned rowU = rowBytes >> 3;      // lengths in 8-byte units
  unsigned strU = strideBytes >> 3;
  u32x4 g0;
  g0[0] = 1u;                                   // count=1 (valid user descriptor)
  g0[1] = ldsAddr;                              // lds_addr (bytes)
  g0[2] = (unsigned)ga;                         // global_addr[31:0]
  g0[3] = (unsigned)(ga >> 32) | (2u << 30);    // global_addr[56:32] | type=2
  u32x8 g1;
  g1[0] = 3u << 16;                             // workgroup_mask=0, data_size=3 (8B)
  g1[1] = (rowU & 0xFFFFu) << 16;               // tensor_dim0[15:0]
  g1[2] = (rowU >> 16) | (16u << 16);           // tensor_dim0[31:16] | tensor_dim1=16
  g1[3] = (rowU & 0xFFFFu) << 16;               // tensor_dim1[31:16]=0 | tile_dim0=rowU
  g1[4] = 16u;                                  // tile_dim1=16 rows, tile_dim2=0
  g1[5] = strU;                                 // tensor_dim0_stride[31:0]
  g1[6] = 0u;                                   // stride[47:32]=0, dim1_stride[15:0]=0
  g1[7] = 0u;
  asm volatile("tensor_load_to_lds %0, %1" :: "s"(g0), "s"(g1) : "memory");
}

// wave-level GEMM: 16(M) x 64(N) tile; A resident in LDS (row-major, lda elems),
// B fragment-packed in global (L2-resident weights).
__device__ __forceinline__ void gemm16x64(const __bf16* Alds, int lda,
                                          const u32x4* Bp, int ksteps,
                                          int lane, v8f acc[4]) {
  for (int ks = 0; ks < ksteps; ++ks) {
    AFrag a = load_a(Alds + ks * 32, lda, lane);
#pragma unroll
    for (int nt = 0; nt < 4; ++nt) {
      BFrag b;
      const u32x4* p = Bp + ((size_t)nt * ksteps + ks) * 64 + lane * 2;
      b.q[0] = p[0];
      b.q[1] = p[1];
      acc[nt] = wmma_bf16(a, b, acc[nt]);
    }
  }
}

// ---- weight re-pack: fp32 row-major -> bf16 WMMA-B-fragment-major ----------
__global__ __launch_bounds__(256) void pack_weights(
    const float* __restrict__ Wq, const float* __restrict__ Wk,
    const float* __restrict__ Wv, const float* __restrict__ Wo,
    const float* __restrict__ W1, const float* __restrict__ W2,
    __bf16* __restrict__ pq, __bf16* __restrict__ pk, __bf16* __restrict__ pv,
    __bf16* __restrict__ po, __bf16* __restrict__ p1, __bf16* __restrict__ p2) {
  int tid = blockIdx.x * 256 + threadIdx.x;
  const float* src; __bf16* dst;
  int Nn, KS, qkvmode, base;
  if      (tid < 4096)  { src = Wq; dst = pq; Nn = 256;  KS = 8;  qkvmode = 1; base = 0;     }
  else if (tid < 8192)  { src = Wk; dst = pk; Nn = 256;  KS = 8;  qkvmode = 1; base = 4096;  }
  else if (tid < 12288) { src = Wv; dst = pv; Nn = 256;  KS = 8;  qkvmode = 1; base = 8192;  }
  else if (tid < 16384) { src = Wo; dst = po; Nn = 256;  KS = 8;  qkvmode = 0; base = 12288; }
  else if (tid < 32768) { src = W1; dst = p1; Nn = 1024; KS = 8;  qkvmode = 0; base = 16384; }
  else if (tid < 49152) { src = W2; dst = p2; Nn = 256;  KS = 32; qkvmode = 0; base = 32768; }
  else return;
  int idx  = tid - base;
  int lane = idx & 31, frag = idx >> 5;
  int ntile = frag / KS, kstep = frag % KS;
  int half = lane >> 4, col = lane & 15;
  int n = ntile * 16 + col;
#pragma unroll
  for (int vv = 0; vv < 8; ++vv)
#pragma unroll
    for (int pp = 0; pp < 2; ++pp) {
      int kk = kstep * 32 + 2 * vv + pp + 16 * half;
      float val = qkvmode
          ? src[(size_t)(n >> 6) * (C_ * HS) + (size_t)kk * HS + (n & 63)]
          : src[(size_t)kk * Nn + n];
      dst[(size_t)frag * 512 + lane * 16 + vv * 2 + pp] = (__bf16)val;
    }
}

// ---- LayerNorm (one row of C=256 per 256-thread block) ---------------------
__global__ __launch_bounds__(256) void ln_kernel(const float* __restrict__ x,
                                                 const float* __restrict__ g,
                                                 const float* __restrict__ be,
                                                 __bf16* __restrict__ out) {
  __shared__ float red[256];
  int row = blockIdx.x, t = threadIdx.x;
  float v = x[(size_t)row * C_ + t];
  red[t] = v;
  __syncthreads();
  for (int s = 128; s > 0; s >>= 1) { if (t < s) red[t] += red[t + s]; __syncthreads(); }
  float mu = red[0] * (1.0f / C_);
  __syncthreads();
  float d = v - mu;
  red[t] = d * d;
  __syncthreads();
  for (int s = 128; s > 0; s >>= 1) { if (t < s) red[t] += red[t + s]; __syncthreads(); }
  float var = red[0] * (1.0f / C_);
  float r = rsqrtf(var + 1e-5f);
  out[(size_t)row * C_ + t] = (__bf16)(d * r * g[t] + be[t]);
}

// ---- QKV projection (q scaled by 1/sqrt(hs); v stored transposed) ----------
__global__ __launch_bounds__(128) void qkv_kernel(const __bf16* __restrict__ h1,
    const u32x4* __restrict__ pq, const u32x4* __restrict__ pk,
    const u32x4* __restrict__ pv,
    __bf16* __restrict__ q, __bf16* __restrict__ kmat, __bf16* __restrict__ vT) {
  __shared__ __bf16 tileA[16 * C_];
  int lane = threadIdx.x & 31;
  int wv   = blockIdx.x * 4 + (threadIdx.x >> 5);
  int mat  = wv / 6144;               // 0=q, 1=k, 2=v   (uniform per block)
  int r    = wv % 6144;
  int b    = r / 192;
  int r2   = r % 192;
  int mt   = r2 / 4, ns = r2 % 4;     // mt uniform per block; ns == head
  const __bf16* A  = h1 + ((size_t)b * T_ + mt * 16) * C_;
  const u32x4*  Bp = (mat == 0 ? pq : (mat == 1 ? pk : pv)) + (size_t)ns * 4 * 8 * 64;

  if ((threadIdx.x >> 5) == 0) {      // one TDM issue per block
    tdm_load_tile16(A, (unsigned)(size_t)&tileA[0], C_ * 2, C_ * 2);
    __builtin_amdgcn_s_wait_tensorcnt(0);
  }
  __syncthreads();

  v8f zero = {0, 0, 0, 0, 0, 0, 0, 0};
  v8f acc[4] = {zero, zero, zero, zero};
  gemm16x64(&tileA[0], C_, Bp, 8, lane, acc);
  int half = lane >> 4, col = lane & 15;
  float scale = (mat == 0) ? 0.125f : 1.0f;   // 1/sqrt(64) folded into q
#pragma unroll
  for (int nt = 0; nt < 4; ++nt)
#pragma unroll
    for (int i = 0; i < 8; ++i) {
      int m = mt * 16 + i + 8 * half;
      int d = nt * 16 + col;
      float val = acc[nt][i] * scale;
      if (mat == 2)
        vT[((size_t)(b * H_ + ns) * HS + d) * T_ + m] = (__bf16)val;
      else {
        __bf16* dst = (mat == 0) ? q : kmat;
        dst[((size_t)(b * H_ + ns) * T_ + m) * HS + d] = (__bf16)val;
      }
    }
}

// ---- Flash attention: wave = 16 query rows, causal 32-key blocks -----------
__global__ __launch_bounds__(128) void attn_kernel(const __bf16* __restrict__ q,
    const __bf16* __restrict__ k, const __bf16* __restrict__ vT,
    __bf16* __restrict__ o) {
  __shared__ __bf16 plds[4][16 * 32];
  int lane = threadIdx.x & 31, warp = threadIdx.x >> 5;
  int wv = blockIdx.x * 4 + warp;
  int b  = wv / (H_ * 48);
  int r  = wv % (H_ * 48);
  int h  = r / 48;
  int mt = r % 48;
  int qbase = mt * 16;
  int half = lane >> 4, col = lane & 15;

  const __bf16* qp = q + ((size_t)(b * H_ + h) * T_ + qbase) * HS;
  AFrag qa0 = load_a(qp, HS, lane);        // d = 0..31
  AFrag qa1 = load_a(qp + 32, HS, lane);   // d = 32..63
  const __bf16* kp = k  + (size_t)(b * H_ + h) * T_ * HS;
  const __bf16* vp = vT + (size_t)(b * H_ + h) * HS * T_;

  float rm[8], rl[8];
#pragma unroll
  for (int i = 0; i < 8; ++i) { rm[i] = -3.0e38f; rl[i] = 0.0f; }
  v8f zero = {0, 0, 0, 0, 0, 0, 0, 0};
  v8f oacc[4] = {zero, zero, zero, zero};
  __bf16* pl = &plds[warp][0];

  for (int s0 = 0; s0 < qbase + 16; s0 += 32) {
    float p[2][8];
#pragma unroll
    for (int sub = 0; sub < 2; ++sub) {
      int kbk = s0 + sub * 16;
      if (kbk < qbase + 16) {                 // uniform per wave
        BFrag k0 = load_bT(kp + (size_t)kbk * HS,      HS, lane);
        BFrag k1 = load_bT(kp + (size_t)kbk * HS + 32, HS, lane);
        v8f s = wmma_bf16(qa0, k0, zero);
        s = wmma_bf16(qa1, k1, s);
#pragma unroll
        for (int i = 0; i < 8; ++i) {
          int tq = qbase + i + 8 * half;
          int sk = kbk + col;
          p[sub][i] = (sk <= tq) ? s[i] : -3.0e38f;   // causal mask
        }
      } else {
#pragma unroll
        for (int i = 0; i < 8; ++i) p[sub][i] = -3.0e38f;
      }
    }
    // row-wise max over 32 keys (16 lanes of this half hold the 16 cols)
    float mb[8];
#pragma unroll
    for (int i = 0; i < 8; ++i) mb[i] = fmaxf(p[0][i], p[1][i]);
#pragma unroll
    for (int off = 1; off < 16; off <<= 1)
#pragma unroll
      for (int i = 0; i < 8; ++i)
        mb[i] = fmaxf(mb[i], __shfl_xor(mb[i], off, 32));
    float sc[8], rs[8];
#pragma unroll
    for (int i = 0; i < 8; ++i) {
      float mn = fmaxf(rm[i], mb[i]);
      sc[i] = __expf(rm[i] - mn);
      rm[i] = mn;
      p[0][i] = __expf(p[0][i] - mn);
      p[1][i] = __expf(p[1][i] - mn);
      rs[i] = p[0][i] + p[1][i];
    }
#pragma unroll
    for (int off = 1; off < 16; off <<= 1)
#pragma unroll
      for (int i = 0; i < 8; ++i)
        rs[i] += __shfl_xor(rs[i], off, 32);
#pragma unroll
    for (int i = 0; i < 8; ++i) rl[i] = rl[i] * sc[i] + rs[i];
#pragma unroll
    for (int nt = 0; nt < 4; ++nt)
#pragma unroll
      for (int i = 0; i < 8; ++i) oacc[nt][i] = oacc[nt][i] * sc[i];

    // P (C/D layout) -> LDS (row-major bf16) -> reload as A fragment
#pragma unroll
    for (int sub = 0; sub < 2; ++sub)
#pragma unroll
      for (int i = 0; i < 8; ++i)
        pl[(i + 8 * half) * 32 + sub * 16 + col] = (__bf16)p[sub][i];
    asm volatile("s_wait_dscnt 0" ::: "memory");
    AFrag pa = load_a(pl, 32, lane);
#pragma unroll
    for (int nt = 0; nt < 4; ++nt) {
      BFrag vf = load_bT(vp + (size_t)(nt * 16) * T_ + s0, T_, lane);
      oacc[nt] = wmma_bf16(pa, vf, oacc[nt]);
    }
  }
  float inv[8];
#pragma unroll
  for (int i = 0; i < 8; ++i) inv[i] = 1.0f / rl[i];
#pragma unroll
  for (int nt = 0; nt < 4; ++nt)
#pragma unroll
    for (int i = 0; i < 8; ++i) {
      size_t row = (size_t)b * T_ + qbase + i + 8 * half;
      o[row * C_ + h * HS + nt * 16 + col] = (__bf16)(oacc[nt][i] * inv[i]);
    }
}

// ---- output projection + residual ------------------------------------------
__global__ __launch_bounds__(128) void proj_kernel(const __bf16* __restrict__ ob,
    const u32x4* __restrict__ po, const float* __restrict__ x,
    const float* __restrict__ bo, float* __restrict__ x2) {
  __shared__ __bf16 tileA[16 * C_];
  int lane = threadIdx.x & 31;
  int wv = blockIdx.x * 4 + (threadIdx.x >> 5);
  int mt = wv / 4, ns = wv % 4;       // mt uniform per block
  const __bf16* A  = ob + (size_t)mt * 16 * C_;
  const u32x4*  Bp = po + (size_t)ns * 4 * 8 * 64;

  if ((threadIdx.x >> 5) == 0) {
    tdm_load_tile16(A, (unsigned)(size_t)&tileA[0], C_ * 2, C_ * 2);
    __builtin_amdgcn_s_wait_tensorcnt(0);
  }
  __syncthreads();

  v8f zero = {0, 0, 0, 0, 0, 0, 0, 0};
  v8f acc[4] = {zero, zero, zero, zero};
  gemm16x64(&tileA[0], C_, Bp, 8, lane, acc);
  int half = lane >> 4, col = lane & 15;
#pragma unroll
  for (int nt = 0; nt < 4; ++nt)
#pragma unroll
    for (int i = 0; i < 8; ++i) {
      size_t m = (size_t)mt * 16 + i + 8 * half;
      int n = ns * 64 + nt * 16 + col;
      x2[m * C_ + n] = x[m * C_ + n] + acc[nt][i] + bo[n];
    }
}

// ---- FFN up-projection + ReLU ----------------------------------------------
__global__ __launch_bounds__(128) void ffn1_kernel(const __bf16* __restrict__ h2,
    const u32x4* __restrict__ p1, const float* __restrict__ b1,
    __bf16* __restrict__ a1) {
  __shared__ __bf16 tileA[16 * C_];
  int lane = threadIdx.x & 31;
  int wv = blockIdx.x * 4 + (threadIdx.x >> 5);
  int mt = wv / 16, ns = wv % 16;     // mt uniform per block (4 consecutive ns)
  const __bf16* A  = h2 + (size_t)mt * 16 * C_;
  const u32x4*  Bp = p1 + (size_t)ns * 4 * 8 * 64;

  if ((threadIdx.x >> 5) == 0) {
    tdm_load_tile16(A, (unsigned)(size_t)&tileA[0], C_ * 2, C_ * 2);
    __builtin_amdgcn_s_wait_tensorcnt(0);
  }
  __syncthreads();

  v8f zero = {0, 0, 0, 0, 0, 0, 0, 0};
  v8f acc[4] = {zero, zero, zero, zero};
  gemm16x64(&tileA[0], C_, Bp, 8, lane, acc);
  int half = lane >> 4, col = lane & 15;
#pragma unroll
  for (int nt = 0; nt < 4; ++nt)
#pragma unroll
    for (int i = 0; i < 8; ++i) {
      size_t m = (size_t)mt * 16 + i + 8 * half;
      int n = ns * 64 + nt * 16 + col;
      a1[m * F_ + n] = (__bf16)fmaxf(acc[nt][i] + b1[n], 0.0f);
    }
}

// ---- FFN down-projection + residual -> final output ------------------------
__global__ __launch_bounds__(128) void ffn2_kernel(const __bf16* __restrict__ a1,
    const u32x4* __restrict__ p2, const float* __restrict__ x2,
    const float* __restrict__ b2, float* __restrict__ out) {
  __shared__ __bf16 tileA[16 * F_];   // 32 KB
  int lane = threadIdx.x & 31;
  int wv = blockIdx.x * 4 + (threadIdx.x >> 5);
  int mt = wv / 4, ns = wv % 4;       // mt uniform per block
  const __bf16* A  = a1 + (size_t)mt * 16 * F_;
  const u32x4*  Bp = p2 + (size_t)ns * 4 * 32 * 64;

  if ((threadIdx.x >> 5) == 0) {
    tdm_load_tile16(A, (unsigned)(size_t)&tileA[0], F_ * 2, F_ * 2);
    __builtin_amdgcn_s_wait_tensorcnt(0);
  }
  __syncthreads();

  v8f zero = {0, 0, 0, 0, 0, 0, 0, 0};
  v8f acc[4] = {zero, zero, zero, zero};
  gemm16x64(&tileA[0], F_, Bp, 32, lane, acc);
  int half = lane >> 4, col = lane & 15;
#pragma unroll
  for (int nt = 0; nt < 4; ++nt)
#pragma unroll
    for (int i = 0; i < 8; ++i) {
      size_t m = (size_t)mt * 16 + i + 8 * half;
      int n = ns * 64 + nt * 16 + col;
      out[m * C_ + n] = x2[m * C_ + n] + acc[nt][i] + b2[n];
    }
}

// ---- host entry ------------------------------------------------------------
extern "C" void kernel_launch(void* const* d_in, const int* in_sizes, int n_in,
                              void* d_out, int out_size, void* d_ws, size_t ws_size,
                              hipStream_t stream) {
  const float* x   = (const float*)d_in[0];
  const float* Wq  = (const float*)d_in[1];
  const float* Wk  = (const float*)d_in[2];
  const float* Wv  = (const float*)d_in[3];
  const float* Wo  = (const float*)d_in[4];
  const float* bo  = (const float*)d_in[5];
  const float* W1  = (const float*)d_in[6];
  const float* b1  = (const float*)d_in[7];
  const float* W2  = (const float*)d_in[8];
  const float* b2  = (const float*)d_in[9];
  const float* g1  = (const float*)d_in[10];
  const float* be1 = (const float*)d_in[11];
  const float* g2  = (const float*)d_in[12];
  const float* be2 = (const float*)d_in[13];
  float* out = (float*)d_out;
  (void)in_sizes; (void)n_in; (void)out_size; (void)ws_size;

  char* w = (char*)d_ws;
  size_t off = 0;
  auto take = [&](size_t bytes) -> void* {
    void* p = w + off;
    off += (bytes + 255) & ~(size_t)255;
    return p;
  };
  __bf16* h1 = (__bf16*)take((size_t)BT_ * C_ * 2);
  __bf16* qb = (__bf16*)take((size_t)BT_ * C_ * 2);
  __bf16* kb = (__bf16*)take((size_t)BT_ * C_ * 2);
  __bf16* vT = (__bf16*)take((size_t)BT_ * C_ * 2);
  __bf16* ob = (__bf16*)take((size_t)BT_ * C_ * 2);
  float*  x2 = (float*) take((size_t)BT_ * C_ * 4);
  __bf16* h2 = (__bf16*)take((size_t)BT_ * C_ * 2);
  __bf16* a1 = (__bf16*)take((size_t)BT_ * F_ * 2);
  __bf16* pq = (__bf16*)take((size_t)C_ * C_ * 2);
  __bf16* pk = (__bf16*)take((size_t)C_ * C_ * 2);
  __bf16* pv = (__bf16*)take((size_t)C_ * C_ * 2);
  __bf16* po = (__bf16*)take((size_t)C_ * C_ * 2);
  __bf16* p1 = (__bf16*)take((size_t)C_ * F_ * 2);
  __bf16* p2 = (__bf16*)take((size_t)F_ * C_ * 2);

  pack_weights<<<192, 256, 0, stream>>>(Wq, Wk, Wv, Wo, W1, W2,
                                        pq, pk, pv, po, p1, p2);
  ln_kernel<<<BT_, 256, 0, stream>>>(x, g1, be1, h1);
  qkv_kernel<<<4608, 128, 0, stream>>>(h1, (const u32x4*)pq, (const u32x4*)pk,
                                       (const u32x4*)pv, qb, kb, vT);
  attn_kernel<<<1536, 128, 0, stream>>>(qb, kb, vT, ob);
  proj_kernel<<<1536, 128, 0, stream>>>(ob, (const u32x4*)po, x, bo, x2);
  ln_kernel<<<BT_, 256, 0, stream>>>(x2, g2, be2, h2);
  ffn1_kernel<<<6144, 128, 0, stream>>>(h2, (const u32x4*)p1, b1, a1);
  ffn2_kernel<<<1536, 128, 0, stream>>>(a1, (const u32x4*)p2, x2, b2, out);
}
